// EmbeddingRefinementLayer_85246510891345
// MI455X (gfx1250) — matrix operand order repaired and checked by reading, not compiled, not run on hardware
//
#include <hip/hip_runtime.h>

typedef __attribute__((ext_vector_type(16))) _Float16 v16h;
typedef __attribute__((ext_vector_type(8)))  _Float16 v8h;
typedef __attribute__((ext_vector_type(8)))  float    v8f;
typedef __attribute__((ext_vector_type(4)))  float    v4f;

#define B_  16
#define S_  8192
#define D_  128
#define H_  256
#define LRc 0.1f
#define EPSc 1e-5f
#define MT  32          // rows per workgroup
#define NT_ 256         // threads per workgroup (8 waves of 32)

// ---- WMMA fragment helpers (f16 16x16x32, f32 accumulate) -------------------

// A-matrix 16x32 f16 (ISA 7.12.2): lane m = lane&15; lanes 0-15 hold K groups
// {kb..kb+7, kb+16..kb+23}, lanes 16-31 hold {kb+8..kb+15, kb+24..kb+31}.
__device__ __forceinline__ v16h load_a_frag(const _Float16* lds, int stride,
                                            int mbase, int kb, int lane) {
    int m = mbase + (lane & 15);
    int g = (lane & 16) ? 8 : 0;
    const _Float16* p = lds + m * stride + kb + g;
    v8h lo = *(const v8h*)p;            // ds_load_b128
    v8h hi = *(const v8h*)(p + 16);     // ds_load_b128
    return __builtin_shufflevector(lo, hi, 0,1,2,3,4,5,6,7,8,9,10,11,12,13,14,15);
}

// B-matrix 32x16 f16: lane column n = lane&15; lanes 0-15 K={kb..kb+7, kb+16..kb+23},
// lanes 16-31 K={kb+8..kb+15, kb+24..kb+31}. Weights stored transposed [N][K] (contiguous K).
__device__ __forceinline__ v16h load_b_frag(const _Float16* wt, int klen,
                                            int n, int kb, int lane) {
    int kh = (lane & 16) ? 8 : 0;
    const _Float16* p = wt + (size_t)n * klen + kb + kh;
    v8h lo = *(const v8h*)p;
    v8h hi = *(const v8h*)(p + 16);
    return __builtin_shufflevector(lo, hi, 0,1,2,3,4,5,6,7,8,9,10,11,12,13,14,15);
}

__device__ __forceinline__ v8f wmma16(v16h a, v16h b, v8f c) {
    return __builtin_amdgcn_wmma_f32_16x16x32_f16(false, a, false, b,
                                                  (short)0, c, false, false);
}

// ---- one-time weight convert + transpose to f16 [N][K] ----------------------
__global__ __launch_bounds__(256)
void prep_weights(const float* __restrict__ W1, const float* __restrict__ W2,
                  const float* __restrict__ Wg1, const float* __restrict__ Wg2,
                  _Float16* __restrict__ W1T, _Float16* __restrict__ W2T,
                  _Float16* __restrict__ Wg1T, _Float16* __restrict__ Wg2T) {
    int idx = blockIdx.x * 256 + threadIdx.x;
    if (idx < 32768) {                       // W1 (128x256) -> W1T [256][128]
        int n = idx >> 7, k = idx & 127;
        W1T[idx] = (_Float16)W1[k * 256 + n];
    } else if (idx < 65536) {                // W2 (256x128) -> W2T [128][256]
        int l = idx - 32768; int n = l >> 8, k = l & 255;
        W2T[l] = (_Float16)W2[k * 128 + n];
    } else if (idx < 131072) {               // Wg1 (256x256) -> Wg1T [256][256]
        int l = idx - 65536; int n = l >> 8, k = l & 255;
        Wg1T[l] = (_Float16)Wg1[k * 256 + n];
    } else if (idx < 163840) {               // Wg2 (256x128) -> Wg2T [128][256]
        int l = idx - 131072; int n = l >> 8, k = l & 255;
        Wg2T[l] = (_Float16)Wg2[k * 128 + n];
    }
}

// ---- fused refinement iteration ---------------------------------------------
__global__ __launch_bounds__(NT_)
void refine_iter(const float* __restrict__ in, const float* __restrict__ emb,
                 float* __restrict__ out,
                 const _Float16* __restrict__ W1T, const _Float16* __restrict__ W2T,
                 const _Float16* __restrict__ Wg1T, const _Float16* __restrict__ Wg2T,
                 const float* __restrict__ b1, const float* __restrict__ ln_g,
                 const float* __restrict__ ln_b, const float* __restrict__ b2,
                 const float* __restrict__ bg1, const float* __restrict__ bg2) {
    __shared__ __align__(16) _Float16 xA[MT * D_];    // 8 KB  f16 refined rows
    __shared__ __align__(16) _Float16 ctxA[MT * D_];  // 8 KB  f16 context rows
    __shared__ __align__(16) _Float16 h1s[MT * H_];   // 16 KB f16 post-LN relu
    __shared__ __align__(16) float    hpre[MT * H_];  // 32 KB f32 pre-LN (reused as g1 f16)
    _Float16* g1s = (_Float16*)hpre;

    const int t    = threadIdx.x;
    const int lane = t & 31;
    const int wave = t >> 5;
    const int r0   = blockIdx.x * MT;      // global row = b*S + s
    const int s0   = r0 & (S_ - 1);        // S power of two; blocks never straddle b

    // Phase 1: stage x and ctx (f32 -> f16) into LDS, vectorized b128 stores
    {
        int row   = t >> 3;
        int dbase = (t & 7) * 16;
        long grow = (long)(r0 + row);
        int  s    = s0 + row;
        long rm, rp;
        if (s == 0)           { rm = grow + 1; rp = grow + 1; }
        else if (s == S_ - 1) { rm = grow - 1; rp = grow - 1; }
        else                  { rm = grow - 1; rp = grow + 1; }
        const float* xr = in + grow * D_ + dbase;
        const float* pm = in + rm   * D_ + dbase;
        const float* pp = in + rp   * D_ + dbase;
        #pragma unroll
        for (int i = 0; i < 16; i += 8) {
            v4f x0 = *(const v4f*)(xr + i),     x1 = *(const v4f*)(xr + i + 4);
            v4f m0 = *(const v4f*)(pm + i),     m1 = *(const v4f*)(pm + i + 4);
            v4f p0 = *(const v4f*)(pp + i),     p1 = *(const v4f*)(pp + i + 4);
            v8h xh, ch;
            #pragma unroll
            for (int j = 0; j < 4; ++j) {
                xh[j]     = (_Float16)x0[j];
                xh[j + 4] = (_Float16)x1[j];
                ch[j]     = (_Float16)(0.5f * (m0[j] + p0[j]));
                ch[j + 4] = (_Float16)(0.5f * (m1[j] + p1[j]));
            }
            *(v8h*)(xA   + row * D_ + dbase + i) = xh;   // ds_store_b128
            *(v8h*)(ctxA + row * D_ + dbase + i) = ch;   // ds_store_b128
        }
    }
    __syncthreads();

    const int mt1  = wave & 1;             // M tile (2 of 16 rows)
    const int ntg  = wave >> 1;            // N tile group
    const int nloc = lane & 15;
    const int hi8  = (lane & 16) ? 8 : 0;

    int nj[4];
    #pragma unroll
    for (int j = 0; j < 4; ++j) nj[j] = (ntg + j * 4) * 16 + nloc;

    // Phase 2: hpre = x @ W1 + b1   (M=32, N=256, K=128)
    // K-outer: one A fragment feeds 4 N-tiles (4x fewer LDS reads)
    {
        v8f acc[4];
        #pragma unroll
        for (int j = 0; j < 4; ++j) {
            float bv = b1[nj[j]];
            acc[j] = (v8f){bv, bv, bv, bv, bv, bv, bv, bv};
        }
        #pragma unroll
        for (int kc = 0; kc < 4; ++kc) {
            v16h a = load_a_frag(xA, D_, mt1 * 16, kc * 32, lane);
            #pragma unroll
            for (int j = 0; j < 4; ++j) {
                v16h b = load_b_frag(W1T, D_, nj[j], kc * 32, lane);
                acc[j] = wmma16(a, b, acc[j]);
            }
        }
        #pragma unroll
        for (int j = 0; j < 4; ++j)
            #pragma unroll
            for (int r = 0; r < 8; ++r)
                hpre[(mt1 * 16 + r + hi8) * H_ + nj[j]] = acc[j][r];
    }
    __syncthreads();

    // Phase 3: LayerNorm over H + relu -> h1s (f16). 8 lanes per row.
    {
        int row = t >> 3, seg = t & 7;
        const float* hr = hpre + row * H_ + seg * 32;
        float s = 0.f, s2 = 0.f;
        #pragma unroll
        for (int i = 0; i < 32; ++i) { float v = hr[i]; s += v; s2 += v * v; }
        #pragma unroll
        for (int off = 1; off < 8; off <<= 1) {
            s  += __shfl_xor(s,  off, 8);
            s2 += __shfl_xor(s2, off, 8);
        }
        float mu  = s  * (1.0f / H_);
        float var = s2 * (1.0f / H_) - mu * mu;
        float rs  = rsqrtf(var + EPSc);
        #pragma unroll
        for (int i = 0; i < 32; ++i) {
            int n = seg * 32 + i;
            float y = (hr[i] - mu) * rs * ln_g[n] + ln_b[n];
            h1s[row * H_ + n] = (_Float16)fmaxf(y, 0.f);
        }
    }
    __syncthreads();

    // Phase 4: g1 = relu([x, ctx] @ Wg1 + bg1)   (K=256); g1s overlays hpre
    {
        v8f acc[4];
        #pragma unroll
        for (int j = 0; j < 4; ++j) {
            float bv = bg1[nj[j]];
            acc[j] = (v8f){bv, bv, bv, bv, bv, bv, bv, bv};
        }
        #pragma unroll
        for (int kc = 0; kc < 8; ++kc) {
            const _Float16* asrc = (kc < 4) ? xA : ctxA;
            v16h a = load_a_frag(asrc, D_, mt1 * 16, (kc & 3) * 32, lane);
            #pragma unroll
            for (int j = 0; j < 4; ++j) {
                v16h b = load_b_frag(Wg1T, 2 * D_, nj[j], kc * 32, lane);
                acc[j] = wmma16(a, b, acc[j]);
            }
        }
        #pragma unroll
        for (int j = 0; j < 4; ++j)
            #pragma unroll
            for (int r = 0; r < 8; ++r)
                g1s[(mt1 * 16 + r + hi8) * H_ + nj[j]] = (_Float16)fmaxf(acc[j][r], 0.f);
    }
    __syncthreads();

    // Phase 5: pred = h1@W2+b2, gate = sigmoid(g1@Wg2+bg2), fused update.
    // K-outer: B fragments shared across both M-tiles (2x fewer global reads)
    {
        const int n2 = wave * 16 + nloc;   // N over D (8 tiles, one per wave)
        const float bP = b2[n2], bG = bg2[n2];
        v8f accP[2], accG[2];
        #pragma unroll
        for (int mt = 0; mt < 2; ++mt) {
            accP[mt] = (v8f){bP, bP, bP, bP, bP, bP, bP, bP};
            accG[mt] = (v8f){bG, bG, bG, bG, bG, bG, bG, bG};
        }
        #pragma unroll
        for (int kc = 0; kc < 8; ++kc) {
            v16h bW = load_b_frag(W2T,  H_, n2, kc * 32, lane);
            v16h bg = load_b_frag(Wg2T, H_, n2, kc * 32, lane);
            #pragma unroll
            for (int mt = 0; mt < 2; ++mt) {
                v16h aH = load_a_frag(h1s, H_, mt * 16, kc * 32, lane);
                accP[mt] = wmma16(aH, bW, accP[mt]);
                v16h aG = load_a_frag(g1s, H_, mt * 16, kc * 32, lane);
                accG[mt] = wmma16(aG, bg, accG[mt]);
            }
        }
        #pragma unroll
        for (int mt = 0; mt < 2; ++mt) {
            #pragma unroll
            for (int r = 0; r < 8; ++r) {
                int  row = mt * 16 + r + hi8;
                long g   = (long)(r0 + row) * D_ + n2;
                float e    = emb[g];
                float xv   = in[g];                   // fp32 refined (precision)
                float gate = 1.0f / (1.0f + __expf(-accG[mt][r]));
                out[g] = xv + LRc * gate * (e - accP[mt][r]);
            }
        }
    }
}

// ---- host launch -------------------------------------------------------------
extern "C" void kernel_launch(void* const* d_in, const int* in_sizes, int n_in,
                              void* d_out, int out_size, void* d_ws, size_t ws_size,
                              hipStream_t stream) {
    const float* emb  = (const float*)d_in[0];
    const float* W1   = (const float*)d_in[1];
    const float* b1   = (const float*)d_in[2];
    const float* ln_g = (const float*)d_in[3];
    const float* ln_b = (const float*)d_in[4];
    const float* W2   = (const float*)d_in[5];
    const float* b2   = (const float*)d_in[6];
    const float* Wg1  = (const float*)d_in[7];
    const float* bg1  = (const float*)d_in[8];
    const float* Wg2  = (const float*)d_in[9];
    const float* bg2  = (const float*)d_in[10];

    char* ws = (char*)d_ws;
    float*     bufA = (float*)ws;                                    // 64 MB
    float*     bufB = (float*)(ws + (size_t)(B_*S_*D_) * 4);         // 64 MB
    _Float16*  W1T  = (_Float16*)(ws + (size_t)(B_*S_*D_) * 8);
    _Float16*  W2T  = W1T + 32768;
    _Float16*  Wg1T = W2T + 32768;
    _Float16*  Wg2T = Wg1T + 65536;
    float* outF = (float*)d_out;

    prep_weights<<<640, 256, 0, stream>>>(W1, W2, Wg1, Wg2, W1T, W2T, Wg1T, Wg2T);

    const int grid = (B_ * S_) / MT;     // 4096 workgroups per iteration
    const float* src[5] = { emb,  bufA, bufB, bufA, bufB };
    float*       dst[5] = { bufA, bufB, bufA, bufB, outF };
    for (int it = 0; it < 5; ++it) {
        refine_iter<<<grid, NT_, 0, stream>>>(src[it], emb, dst[it],
                                              W1T, W2T, Wg1T, Wg2T,
                                              b1, ln_g, ln_b, b2, bg1, bg2);
    }
}
